// HQQLinear_29721173688947
// MI455X (gfx1250) — matrix operand order repaired and checked by reading, not compiled, not run on hardware
//
#include <hip/hip_runtime.h>

typedef __attribute__((ext_vector_type(16))) _Float16 v16h;
typedef __attribute__((ext_vector_type(8)))  float    v8f;
typedef __attribute__((ext_vector_type(4)))  float    f4;
typedef __attribute__((ext_vector_type(4)))  int      i4;

#define OUT_DIM 11008
#define IN_DIM  4096
#define GNUM    704512     // number of quant groups (columns of W_q)
#define RLEN    172        // W rows covered by one packed row block (704512/4096)
#define NT      688        // 11008 / 16 output tiles
#define KB_TOT  128        // K blocks of 32 (4096/32)
#define WAVES_PER_BLOCK 2

// One wave computes a 16(M=batch, rows 8..15 duplicate rows 0..7) x 16(N=out)
// tile over a K chunk of 4096/SPLIT using v_wmma_f32_16x16x32_f16,
// dequantizing HQQ 4-bit weights on the fly. Traffic floor ~96 MB -> ~4.1 us
// at 23.3 TB/s; SPLIT>1 raises wave count (688*SPLIT) for latency hiding.
template <int SPLIT>
__global__ __launch_bounds__(64) void hqq_wmma_kernel(
    const int*   __restrict__ Wq,     // [32, 704512] int32, one packed byte each
    const float* __restrict__ scale,  // [704512]
    const float* __restrict__ zero,   // [704512]
    const float* __restrict__ x,      // [8, 4096]
    const float* __restrict__ bias,   // [11008]
    float*       __restrict__ out,    // [8, 11008]      (SPLIT == 1)
    float*       __restrict__ part)   // [NT*SPLIT*8*16] (SPLIT > 1)
{
    const int lane  = threadIdx.x & 31;
    const int gw    = blockIdx.x * WAVES_PER_BLOCK + (threadIdx.x >> 5);
    const int tile  = gw / SPLIT;     // output tile (SPLIT is a power of 2)
    const int kc    = gw % SPLIT;     // K chunk

    const int n     = lane & 15;      // B column / D column
    const int khalf = lane >> 4;      // which K half this lane holds
    const int o     = tile * 16 + n;  // output feature index

    // W[o,i] lives at packed row (o/172) (hi nibble if <32 else lo of row-32),
    // group column g = (o%172)*4096 + i  -> contiguous in i per output row.
    const int   r     = o / RLEN;                  // 0..63
    const int   row_q = (r < 32) ? r : (r - 32);   // packed W_q row
    const int   imask = (r < 32) ? 0xF0 : 0x0F;    // nibble select by masking
    const float ks    = (r < 32) ? 0.0625f : 1.0f; // fold >>4 into the scale
    const int   g0    = (o % RLEN) * IN_DIM;

    const int*   wq_p = Wq + (size_t)row_q * GNUM + g0;
    const float* sc_p = scale + g0;
    const float* zr_p = zero  + g0;

    // A row = batch index. Rows 8..15 duplicate rows 0..7: their D rows are
    // discarded in the epilogue, so no zero-padding (no divergence) needed.
    const float* x_p = x + (size_t)(n & 7) * IN_DIM;

    v8f acc = {};

    const int kb0 = kc * (KB_TOT / SPLIT);
    for (int kb = kb0; kb < kb0 + KB_TOT / SPLIT; ++kb) {
        const int ib = kb * 32 + khalf * 16;  // 16 consecutive i for B operand

        // Prefetch the dominant W_q stream ~8 iterations ahead. Lowered as
        // TH=0 speculative (translation faults silently dropped), so small
        // end-of-buffer overruns are safe.
        __builtin_prefetch(wq_p + ib + 256, 0, 0);

        // ---- B operand: load 16 packed ints + 16 scales + 16 zeros ----
        i4 qv[4]; f4 sv[4]; f4 zv[4];
#pragma unroll
        for (int j = 0; j < 4; ++j) {
            qv[j] = *(const i4*)(wq_p + ib + 4 * j);
            sv[j] = *(const f4*)(sc_p + ib + 4 * j);
            zv[j] = *(const f4*)(zr_p + ib + 4 * j);
        }
        const int*   qa = (const int*)qv;
        const float* sa = (const float*)sv;
        const float* za = (const float*)zv;

        // Dequantize: w = (cvt(v & mask) * ks - zero) * scale  -> f16
        v16h bf;
#pragma unroll
        for (int e = 0; e < 16; ++e) {
            float qf = (float)(qa[e] & imask);
            bf[e] = (_Float16)((qf * ks - za[e]) * sa[e]);
        }

        // ---- A operand ----
        // A-layout matching B: lanes hold K = khalf*16+{0..15} (contiguous).
        const float* xp = x_p + kb * 32 + khalf * 16;
        f4 xa[4];
        xa[0] = *(const f4*)(xp);
        xa[1] = *(const f4*)(xp + 4);
        xa[2] = *(const f4*)(xp + 8);
        xa[3] = *(const f4*)(xp + 12);
        const float* xf = (const float*)xa;
        v16h af;
#pragma unroll
        for (int e = 0; e < 16; ++e)
            af[e] = (_Float16)xf[e];

        // D = A x B + C  (16x16x32, f16 in / f32 accumulate)
        acc = __builtin_amdgcn_wmma_f32_16x16x32_f16(
            /*neg_a=*/false, af, /*neg_b=*/false, bf,
            /*c_mod=*/(short)0, acc, /*reuse_a=*/false, /*reuse_b=*/false);
    }

    // D layout: lanes 0-15 hold N=lane, VGPR j = batch row j (rows 8..15 in
    // lanes 16-31 are duplicates -> dropped).
    if (khalf == 0) {
        if (SPLIT == 1) {
            const float bv = bias[o];
            float* op = out + o;
#pragma unroll
            for (int j = 0; j < 8; ++j)
                op[(size_t)j * OUT_DIM] = acc[j] + bv;
        } else {
            float* pp = part + (((size_t)tile * SPLIT + kc) * 8) * 16 + n;
#pragma unroll
            for (int j = 0; j < 8; ++j)
                pp[j * 16] = acc[j];
        }
    }
}

// Deterministic cross-chunk reduction: out[b,o] = bias[o] + sum_kc part[...].
__global__ __launch_bounds__(256) void hqq_reduce_kernel(
    const float* __restrict__ part, const float* __restrict__ bias,
    float* __restrict__ out, int split)
{
    int idx = blockIdx.x * 256 + threadIdx.x;
    if (idx >= 8 * OUT_DIM) return;
    const int b = idx / OUT_DIM;
    const int o = idx - b * OUT_DIM;
    const int tile = o >> 4;
    const int n    = o & 15;
    float sum = bias[o];
    for (int kc = 0; kc < split; ++kc)
        sum += part[(((size_t)tile * split + kc) * 8 + b) * 16 + n];
    out[idx] = sum;
}

extern "C" void kernel_launch(void* const* d_in, const int* in_sizes, int n_in,
                              void* d_out, int out_size, void* d_ws, size_t ws_size,
                              hipStream_t stream) {
    const int*   Wq    = (const int*)  d_in[0];   // [32, 704512] int32
    const float* scale = (const float*)d_in[1];   // [1, 704512]
    const float* zero  = (const float*)d_in[2];   // [1, 704512]
    const float* x     = (const float*)d_in[3];   // [8, 1, 4096]
    const float* bias  = (const float*)d_in[4];   // [11008]
    float*       out   = (float*)d_out;           // [8, 1, 11008]
    float*       part  = (float*)d_ws;

    (void)in_sizes; (void)n_in; (void)out_size;

    const size_t need4 = (size_t)NT * 4 * 8 * 16 * sizeof(float); // ~1.41 MB
    const size_t need2 = need4 / 2;
    const dim3 block(64);
    const dim3 rblock(256), rgrid((8 * OUT_DIM + 255) / 256);

    if (ws_size >= need4) {
        // SPLIT=4: 2752 waves -> good latency hiding, +1.4 MB partial traffic.
        hqq_wmma_kernel<4><<<dim3(NT * 4 / WAVES_PER_BLOCK), block, 0, stream>>>(
            Wq, scale, zero, x, bias, out, part);
        hqq_reduce_kernel<<<rgrid, rblock, 0, stream>>>(part, bias, out, 4);
    } else if (ws_size >= need2) {
        hqq_wmma_kernel<2><<<dim3(NT * 2 / WAVES_PER_BLOCK), block, 0, stream>>>(
            Wq, scale, zero, x, bias, out, part);
        hqq_reduce_kernel<<<rgrid, rblock, 0, stream>>>(part, bias, out, 2);
    } else {
        // Fallback: single pass, bias fused, no workspace needed.
        hqq_wmma_kernel<1><<<dim3(NT / WAVES_PER_BLOCK), block, 0, stream>>>(
            Wq, scale, zero, x, bias, out, part);
    }
}
